// CurriculumLSTMModel_31473520345466
// MI455X (gfx1250) — matrix operand order repaired and checked by reading, not compiled, not run on hardware
//
#include <hip/hip_runtime.h>
#include <hip/hip_bf16.h>
#include <math.h>

typedef __attribute__((ext_vector_type(16))) _Float16 v16h;
typedef __attribute__((ext_vector_type(8)))  float    v8f;

#define NPIX 16384   // 128*128
#define CHUNK 4      // frames per conv pass

__device__ __forceinline__ float sigm(float x) { return 1.0f / (1.0f + __expf(-x)); }

// ---------------------------------------------------------------------------
// Implicit-GEMM 3x3 SAME conv + bias + ReLU, f16 WMMA with f32 accumulate.
// GEMM view: M = Cout, N = pixels, K decomposed as tap(9) x ci-chunk(32) so
// im2col needs no per-element division (kh/kw are loop-uniform scalars).
// Block = 256 threads (8 waves), output tile 64 co x 128 px:
//   4 co-tiles x 8 px-tiles = 32 WMMA tiles; wave w owns co-tile ct=w>>1 and
//   px-tiles (w&1)*4 .. +3  -> 4 accumulators, A-fragment reused 4x.
// LDS tiles stored pre-swizzled in WMMA fragment order:
//   A frag (16x32, MxK): lane L, half h -> m=L&15, k=(h&7)+((h>>3)<<4)+(L>>4)*8
//   B frag (32x16, KxN): lane L, half h -> n=L&15, k=(h&7)+((h>>3)<<4)+(L>>4)*8
//   C/D  (16x16 f32)   : vgpr r, lane L -> m=r+(L>>4)*8, n=L&15
// ---------------------------------------------------------------------------
__global__ __launch_bounds__(256) void conv_wmma(
    const float* __restrict__ x,    // [frame][Cin][128][128] (chunk base)
    const float* __restrict__ wgt,  // [Cout][Cin*9]  (k = ci*9 + kh*3 + kw)
    const float* __restrict__ bias, // [Cout]
    float* __restrict__ y,          // [frame][Cout][128][128]
    int Cin, int Cout)
{
    const int K = Cin * 9;
    const int frame   = blockIdx.z;
    const int coBase  = blockIdx.y * 64;
    const int pixBase = blockIdx.x * 128;
    const float* xin  = x + (size_t)frame * Cin  * NPIX;
    float*       yout = y + (size_t)frame * Cout * NPIX;

    __shared__ __align__(32) _Float16 Alds[4][512];  // 4 co-tiles, frag order
    __shared__ __align__(32) _Float16 Blds[8][512];  // 8 px-tiles, frag order

    const int tid    = threadIdx.x;
    const int lane   = tid & 31;
    const int wave   = tid >> 5;
    const int ct     = wave >> 1;        // co-tile of this wave (0..3)
    const int ptBase = (wave & 1) * 4;   // first px-tile of this wave

    v8f accs[4] = {{}, {}, {}, {}};

    for (int tap = 0; tap < 9; ++tap) {
        const int kh = tap / 3 - 1;      // loop-uniform (SALU)
        const int kw = tap % 3 - 1;
        for (int ci0 = 0; ci0 < Cin; ci0 += 32) {
            __syncthreads();  // previous iteration's consumers are done

            // ---- stage A (weights): 64 co x 32 ci = 2048 elems ----
            for (int e = tid; e < 2048; e += 256) {
                int coL = e >> 5;          // 0..63
                int kl  = e & 31;          // ci within chunk
                int co  = coBase + coL;
                int ci  = ci0 + kl;
                float v = (co < Cout && ci < Cin)
                            ? wgt[(size_t)co * K + ci * 9 + tap] : 0.0f;
                int g  = (kl >> 3) & 1;
                int h  = (kl & 7) + ((kl >> 4) << 3);
                int ln = (coL & 15) + g * 16;
                Alds[coL >> 4][ln * 16 + h] = (_Float16)v;
            }
            // ---- stage B (shifted input): 8 tiles x 16 px x 32 ci = 4096 ----
            for (int e = tid; e < 4096; e += 256) {
                int ptL = e >> 9;          // 0..7
                int n   = (e >> 5) & 15;   // pixel within tile
                int kl  = e & 31;          // ci within chunk
                int ci  = ci0 + kl;
                int p   = pixBase + ptL * 16 + n;
                int hh  = (p >> 7) + kh;
                int ww  = (p & 127) + kw;
                float v = 0.0f;
                if (ci < Cin && hh >= 0 && hh < 128 && ww >= 0 && ww < 128)
                    v = xin[(size_t)ci * NPIX + (hh << 7) + ww];
                int ln = n + (((kl >> 3) & 1) << 4);
                Blds[ptL][ln * 16 + (kl & 7) + ((kl >> 4) << 3)] = (_Float16)v;
            }
            __syncthreads();

            v16h a = *(const v16h*)&Alds[ct][lane * 16];
#pragma unroll
            for (int j = 0; j < 4; ++j) {
                v16h b = *(const v16h*)&Blds[ptBase + j][lane * 16];
                accs[j] = __builtin_amdgcn_wmma_f32_16x16x32_f16(
                    false, a, false, b, (short)0, accs[j], false, false);
            }
        }
    }

    // epilogue: bias + ReLU, C/D layout
    const int n = lane & 15;
    const int g = lane >> 4;
#pragma unroll
    for (int j = 0; j < 4; ++j) {
        int pixel = pixBase + (ptBase + j) * 16 + n;
#pragma unroll
        for (int r = 0; r < 8; ++r) {
            int co = coBase + ct * 16 + r + g * 8;
            if (co < Cout) {
                float v = accs[j][r] + bias[co];
                yout[(size_t)co * NPIX + pixel] = fmaxf(v, 0.0f);
            }
        }
    }
}

// ---------------------------------------------------------------------------
// Spatial softmax: one block per (frame-in-chunk, channel).
// feat row g (=b*T+t): cols [camOff + 2*ch] = E[x], [camOff + 2*ch + 1] = E[y].
// ---------------------------------------------------------------------------
__global__ __launch_bounds__(256) void spatial_softmax(
    const float* __restrict__ x,   // [frame][16][128][128] (chunk base)
    float* __restrict__ feat,      // [40][64]
    int f0, int camOff)
{
    const int ch = blockIdx.x;
    const int fl = blockIdx.y;
    const int g  = f0 + fl;
    const float* p = x + ((size_t)fl * 16 + ch) * NPIX;

    __shared__ float sm[256], s0[256], s1[256], s2[256];
    const int tid = threadIdx.x;

    float mx = -3.4e38f;
    for (int i = tid; i < NPIX; i += 256) mx = fmaxf(mx, p[i]);
    sm[tid] = mx;
    __syncthreads();
    for (int s = 128; s > 0; s >>= 1) {
        if (tid < s) sm[tid] = fmaxf(sm[tid], sm[tid + s]);
        __syncthreads();
    }
    const float M = sm[0];

    float se = 0.0f, sx = 0.0f, sy = 0.0f;
    for (int i = tid; i < NPIX; i += 256) {
        float e  = __expf(p[i] - M);
        float px = -1.0f + (2.0f / 127.0f) * (float)(i & 127);
        float py = -1.0f + (2.0f / 127.0f) * (float)(i >> 7);
        se += e; sx += e * px; sy += e * py;
    }
    s0[tid] = se; s1[tid] = sx; s2[tid] = sy;
    __syncthreads();
    for (int s = 128; s > 0; s >>= 1) {
        if (tid < s) { s0[tid] += s0[tid + s]; s1[tid] += s1[tid + s]; s2[tid] += s2[tid + s]; }
        __syncthreads();
    }
    if (tid == 0) {
        float inv = 1.0f / s0[0];
        feat[g * 64 + camOff + ch * 2 + 0] = s1[0] * inv;
        feat[g * 64 + camOff + ch * 2 + 1] = s2[0] * inv;
    }
}

// ---------------------------------------------------------------------------
// 6-layer LSTM (HID=512), B=4, T=10, as ONE persistent workgroup (32 waves).
// Gates (4x2048) via f16 WMMA: M=16 (batch padded), N=2048 -> 128 n-tiles.
// Wave w owns n-tiles {w, w+32, w+64, w+96}: gate columns 16w..16w+15 of
// groups i,f,g,o respectively -> cell update is wave-local; c-state lives in
// registers, h-state in LDS.
// ---------------------------------------------------------------------------
__global__ __launch_bounds__(1024) void lstm_kernel(
    const float* __restrict__ states,  // (4,10,6)
    const float* __restrict__ feat,    // (40,64)
    const float* __restrict__ wih0,    // (2048,70)
    const float* __restrict__ whh0,    // (2048,512)
    const float* __restrict__ bih0, const float* __restrict__ bhh0,   // (2048)
    const float* __restrict__ wih_r,   // (5,2048,576)
    const float* __restrict__ whh_r,   // (5,2048,512)
    const float* __restrict__ bih_r, const float* __restrict__ bhh_r, // (5,2048)
    const float* __restrict__ out_w,   // (6,576)
    const float* __restrict__ out_b,   // (6)
    float* __restrict__ out)           // (4,10,6)
{
    __shared__ float hbuf[6][4][512];

    const int tid  = threadIdx.x;
    const int lane = tid & 31;
    const int wave = tid >> 5;
    const int m    = lane & 15;
    const int gg2  = lane >> 4;

    for (int e = tid; e < 6 * 4 * 512; e += 1024) ((float*)hbuf)[e] = 0.0f;
    float creg[6][4];
#pragma unroll
    for (int l = 0; l < 6; ++l)
#pragma unroll
        for (int b = 0; b < 4; ++b) creg[l][b] = 0.0f;
    __syncthreads();

    for (int t = 0; t < 10; ++t) {
#pragma unroll
        for (int l = 0; l < 6; ++l) {
            const float* Wih = (l == 0) ? wih0 : wih_r + (size_t)(l - 1) * 2048 * 576;
            const float* Whh = (l == 0) ? whh0 : whh_r + (size_t)(l - 1) * 2048 * 512;
            const float* Bih = (l == 0) ? bih0 : bih_r + (l - 1) * 2048;
            const float* Bhh = (l == 0) ? bhh0 : bhh_r + (l - 1) * 2048;
            const int Kx = (l == 0) ? 70 : 576;

            v8f accs[4];
#pragma unroll
            for (int q = 0; q < 4; ++q) {
                const int n = q * 512 + wave * 16 + m;  // absolute gate column
                v8f acc = {};
                // x @ Wih^T  (x = [states, img] for l==0; [h[l-1], img] else)
                for (int k0 = 0; k0 < Kx; k0 += 32) {
                    v16h a, b;
#pragma unroll
                    for (int h = 0; h < 16; ++h) {
                        int ka = k0 + (h & 7) + ((h >> 3) << 4) + gg2 * 8;
                        float av = 0.0f;
                        if (m < 4 && ka < Kx) {
                            if (l == 0)
                                av = (ka < 6) ? states[(m * 10 + t) * 6 + ka]
                                              : feat[(m * 10 + t) * 64 + ka - 6];
                            else
                                av = (ka < 512) ? hbuf[l - 1][m][ka]
                                                : feat[(m * 10 + t) * 64 + ka - 512];
                        }
                        a[h] = (_Float16)av;
                        int kb = k0 + (h & 7) + ((h >> 3) << 4) + gg2 * 8;
                        b[h] = (_Float16)((kb < Kx) ? Wih[(size_t)n * Kx + kb] : 0.0f);
                    }
                    acc = __builtin_amdgcn_wmma_f32_16x16x32_f16(false, a, false, b,
                                                                 (short)0, acc, false, false);
                }
                // h @ Whh^T  (previous-step h[l], not yet overwritten)
                for (int k0 = 0; k0 < 512; k0 += 32) {
                    v16h a, b;
#pragma unroll
                    for (int h = 0; h < 16; ++h) {
                        int ka = k0 + (h & 7) + ((h >> 3) << 4) + gg2 * 8;
                        a[h] = (_Float16)((m < 4) ? hbuf[l][m][ka] : 0.0f);
                        int kb = k0 + (h & 7) + ((h >> 3) << 4) + gg2 * 8;
                        b[h] = (_Float16)Whh[(size_t)n * 512 + kb];
                    }
                    acc = __builtin_amdgcn_wmma_f32_16x16x32_f16(false, a, false, b,
                                                                 (short)0, acc, false, false);
                }
                accs[q] = acc;
            }
            __syncthreads();  // all waves done reading hbuf[l] / hbuf[l-1]

            if (lane < 16) {  // C/D rows m=0..7 live in lanes 0..15; batches are r<4
                const int u = wave * 16 + lane;  // hidden unit
                const float bi = Bih[u]        + Bhh[u];
                const float bf = Bih[512 + u]  + Bhh[512 + u];
                const float bg = Bih[1024 + u] + Bhh[1024 + u];
                const float bo = Bih[1536 + u] + Bhh[1536 + u];
#pragma unroll
                for (int b = 0; b < 4; ++b) {
                    float gi = accs[0][b] + bi;
                    float gf = accs[1][b] + bf;
                    float gc = accs[2][b] + bg;
                    float go = accs[3][b] + bo;
                    float c2 = sigm(gf) * creg[l][b] + sigm(gi) * tanhf(gc);
                    creg[l][b] = c2;
                    hbuf[l][b][u] = sigm(go) * tanhf(c2);
                }
            }
            __syncthreads();
        }

        // output head: out[b,t,:] = [h5, img] @ out_w^T + out_b  (tiny)
        if (tid < 24) {
            int b = tid / 6, a = tid % 6;
            float s = out_b[a];
            for (int j = 0; j < 512; ++j) s += hbuf[5][b][j] * out_w[a * 576 + j];
            for (int j = 0; j < 64; ++j)  s += feat[(b * 10 + t) * 64 + j] * out_w[a * 576 + 512 + j];
            out[(b * 10 + t) * 6 + a] = s;
        }
        __syncthreads();
    }
}

// ---------------------------------------------------------------------------
// Host launcher. Input order (setup_inputs dict):
//  0 seq_img_main  1 seq_img_secondary  2 seq_states
//  3..10  m_w0,m_b0,..,m_w3,m_b3    11..18 s_w0,s_b0,..,s_w3,s_b3
//  19 wih0 20 whh0 21 bih0 22 bhh0  23 wih_r 24 whh_r 25 bih_r 26 bhh_r
//  27 out_w 28 out_b
// Workspace: feat (40x64 f32) @0; bufA (CHUNK*128*NPIX f32) @64KB; bufB after.
// ---------------------------------------------------------------------------
extern "C" void kernel_launch(void* const* d_in, const int* in_sizes, int n_in,
                              void* d_out, int out_size, void* d_ws, size_t ws_size,
                              hipStream_t stream)
{
    (void)in_sizes; (void)n_in; (void)out_size; (void)ws_size;

    const float* img[2]  = {(const float*)d_in[0], (const float*)d_in[1]};
    const float* states  = (const float*)d_in[2];

    float* feat = (float*)d_ws;
    float* bufA = (float*)((char*)d_ws + 65536);
    float* bufB = bufA + (size_t)CHUNK * 128 * NPIX;   // bufA sized for 128 ch

    for (int cam = 0; cam < 2; ++cam) {
        const float* w[4];
        const float* bs[4];
        for (int j = 0; j < 4; ++j) {
            w[j]  = (const float*)d_in[3 + cam * 8 + 2 * j];
            bs[j] = (const float*)d_in[3 + cam * 8 + 2 * j + 1];
        }
        for (int f0 = 0; f0 < 40; f0 += CHUNK) {
            // grid.x = 16384/128 px-blocks, grid.y = ceil(Cout/64)
            conv_wmma<<<dim3(128, 1, CHUNK), 256, 0, stream>>>(
                img[cam] + (size_t)f0 * 3 * NPIX, w[0], bs[0], bufA, 3, 32);
            conv_wmma<<<dim3(128, 1, CHUNK), 256, 0, stream>>>(
                bufA, w[1], bs[1], bufB, 32, 64);
            conv_wmma<<<dim3(128, 2, CHUNK), 256, 0, stream>>>(
                bufB, w[2], bs[2], bufA, 64, 128);
            conv_wmma<<<dim3(128, 1, CHUNK), 256, 0, stream>>>(
                bufA, w[3], bs[3], bufB, 128, 16);
            spatial_softmax<<<dim3(16, CHUNK), 256, 0, stream>>>(
                bufB, feat, f0, cam * 32);
        }
    }

    lstm_kernel<<<1, 1024, 0, stream>>>(
        states, feat,
        (const float*)d_in[19], (const float*)d_in[20],
        (const float*)d_in[21], (const float*)d_in[22],
        (const float*)d_in[23], (const float*)d_in[24],
        (const float*)d_in[25], (const float*)d_in[26],
        (const float*)d_in[27], (const float*)d_in[28],
        (float*)d_out);
}
